// SelfAttention_36893769072963
// MI455X (gfx1250) — matrix operand order repaired and checked by reading, not compiled, not run on hardware
//
#include <hip/hip_runtime.h>
#include <hip/hip_bf16.h>

// ---------------------------------------------------------------------------
// Problem constants (from reference): B=2, S=2048, DIM=2048, H=16, D=128
// ---------------------------------------------------------------------------
#define Bz    2
#define Sq    2048
#define DIMe  2048
#define NH    16
#define HD    128

typedef __attribute__((ext_vector_type(16))) __bf16 v16bf;
typedef __attribute__((ext_vector_type(8)))  float  v8f;

// Native convert (single v_cvt op, RNE) instead of manual bit twiddling.
__device__ __forceinline__ __bf16 f2bf(float f) { return (__bf16)f; }

// ---------------------------------------------------------------------------
// Kernel 1: P = X * W^T  (q[b,s,e] = sum_d x[b,s,d]*W[e,d]), 64x64 tile per
// block, bf16 WMMA over K=32 chunks. RoPE applied in epilogue for Q/K via
// lane-pair shuffle. Output stored bf16 in [b][h][s][d] layout.
// grid = (DIM/64, B*S/64, 3)   block = 128 (4 waves)
// ---------------------------------------------------------------------------
__global__ __launch_bounds__(128)
void qkv_rope_kernel(const float* __restrict__ x,
                     const float* __restrict__ Wq,
                     const float* __restrict__ Wk,
                     const float* __restrict__ Wv,
                     const float* __restrict__ freqs,
                     __bf16* __restrict__ qo,
                     __bf16* __restrict__ ko,
                     __bf16* __restrict__ vo) {
    __shared__ __bf16 As[64][36];   // X tile   (M x K), +4 pad
    __shared__ __bf16 Bs[64][36];   // W tile   (N x K), +4 pad

    const int tid  = threadIdx.x;
    const int wave = tid >> 5;
    const int lane = tid & 31;
    const int hi   = lane >> 4;      // 0: lanes 0-15, 1: lanes 16-31
    const int ln   = lane & 15;

    const int m0 = blockIdx.y * 64;           // row tile in (B*S)
    const int n0 = blockIdx.x * 64;           // col tile in DIM
    const int z  = blockIdx.z;                // 0=Q 1=K 2=V
    const float* W = (z == 0) ? Wq : (z == 1) ? Wk : Wv;
    __bf16* out    = (z == 0) ? qo : (z == 1) ? ko : vo;
    const bool rope = (z < 2);

    v8f acc[4];
    #pragma unroll
    for (int ns = 0; ns < 4; ++ns)
        acc[ns] = (v8f){0.f,0.f,0.f,0.f,0.f,0.f,0.f,0.f};

    for (int kc = 0; kc < DIMe / 32; ++kc) {
        const int k0 = kc * 32;
        // ---- stage 64x32 A and B tiles, f32 -> bf16 (native cvt) ----
        #pragma unroll
        for (int j = 0; j < 4; ++j) {
            int linear = (j * 128 + tid) * 4;       // 0..2047, step 4
            int row = linear >> 5, col = linear & 31;
            float4 av = *(const float4*)(x + (size_t)(m0 + row) * DIMe + k0 + col);
            As[row][col + 0] = f2bf(av.x);  As[row][col + 1] = f2bf(av.y);
            As[row][col + 2] = f2bf(av.z);  As[row][col + 3] = f2bf(av.w);
            float4 bv = *(const float4*)(W + (size_t)(n0 + row) * DIMe + k0 + col);
            Bs[row][col + 0] = f2bf(bv.x);  Bs[row][col + 1] = f2bf(bv.y);
            Bs[row][col + 2] = f2bf(bv.z);  Bs[row][col + 3] = f2bf(bv.w);
        }
        __syncthreads();

        // ---- A fragment (16x32 bf16, ISA 7.12.2 layout) ----
        v16bf a;
        #pragma unroll
        for (int hh = 0; hh < 16; ++hh) {
            int v  = hh >> 1;
            int kk = ((v < 4) ? 0 : 16) + (hi ? 8 : 0) + ((v & 3) << 1) + (hh & 1);
            a[hh] = As[wave * 16 + ln][kk];
        }
        // ---- 4 N-subtiles: B fragment (32x16) + WMMA ----
        #pragma unroll
        for (int ns = 0; ns < 4; ++ns) {
            v16bf bm;
            #pragma unroll
            for (int hh = 0; hh < 16; ++hh)
                bm[hh] = Bs[ns * 16 + ln][((hh & 8) << 1) + (hi ? 8 : 0) + (hh & 7)];
            acc[ns] = __builtin_amdgcn_wmma_f32_16x16x32_bf16(
                false, a, false, bm, (short)0, acc[ns], false, false);
        }
        __syncthreads();
    }

    // ---- epilogue: RoPE (Q,K) + store bf16 [b][h][s][d] ----
    #pragma unroll
    for (int ns = 0; ns < 4; ++ns) {
        const int e    = n0 + ns * 16 + ln;
        const int head = e >> 7;
        const int d    = e & (HD - 1);
        #pragma unroll
        for (int r = 0; r < 8; ++r) {
            const int mg = m0 + wave * 16 + r + (hi ? 8 : 0);
            const int bb = mg >> 11;          // / S
            const int ss = mg & (Sq - 1);     // % S
            float val = acc[ns][r];
            if (rope) {
                float f  = freqs[ss * (HD / 2) + (d >> 1)];
                float c  = __cosf(f), sn = __sinf(f);
                float p  = __shfl_xor(val, 1, 32);   // partner column (d^1)
                val = (d & 1) ? (p * sn + val * c) : (val * c - p * sn);
            }
            out[(((size_t)(bb * NH + head)) * Sq + ss) * HD + d] = f2bf(val);
        }
    }
}

// ---------------------------------------------------------------------------
// Async global->LDS tile copy (CDNA5 GLOBAL_LOAD_ASYNC_TO_LDS_B128).
// Copies a 64x128 bf16 tile (contiguous rows of 128) into LDS with pitch
// LP=136 bf16 (row stride 272 B: 16B-aligned for B128, +4 banks/row stagger).
// No VGPR round-trip; tracked by ASYNCcnt.
// ---------------------------------------------------------------------------
#define LP 136

__device__ __forceinline__
void async_load_tile(const __bf16* __restrict__ src, unsigned lds_base, int tid) {
    #pragma unroll
    for (int j = 0; j < 8; ++j) {
        int linear = j * 128 + tid;        // 1024 x 16-byte chunks
        int row = linear >> 4;
        int c8  = (linear & 15) << 3;
        unsigned loff = lds_base + (unsigned)(row * LP + c8) * 2u;
        const void* g = (const void*)(src + (size_t)row * HD + c8);
        asm volatile("global_load_async_to_lds_b128 %0, %1, off"
                     :: "v"(loff), "v"(g) : "memory");
    }
}

__device__ __forceinline__ void wait_async_all() {
    asm volatile("s_wait_asynccnt 0x0" ::: "memory");
}

__device__ __forceinline__ unsigned lds_off32(const void* p) {
    return (unsigned)(uintptr_t)p;   // low 32 bits of generic addr == LDS offset
}

// ---------------------------------------------------------------------------
// Kernel 2: flash attention per (b, h, 64-row q tile). 4 waves, 16 q rows
// each. Online softmax; causality applied analytically (== -1e9 mask input).
// grid = (S/64, NH, B)   block = 128
// ---------------------------------------------------------------------------
__global__ __launch_bounds__(128)
void flash_attn_kernel(const __bf16* __restrict__ Qg,
                       const __bf16* __restrict__ Kg,
                       const __bf16* __restrict__ Vg,
                       float* __restrict__ out) {
    __shared__ __bf16 Qs[64][LP];
    __shared__ __bf16 Ks[64][LP];
    __shared__ __bf16 Vs[64][LP];
    __shared__ __bf16 Ps[4][16][68];    // per-wave P staging (C-layout -> A-layout)

    const int tid  = threadIdx.x;
    const int wave = tid >> 5;
    const int lane = tid & 31;
    const int hi   = lane >> 4;
    const int ln   = lane & 15;

    const int qt = blockIdx.x;
    const int h  = blockIdx.y;
    const int b  = blockIdx.z;
    const size_t headoff = ((size_t)(b * NH + h)) * Sq * HD;

    const unsigned qbase = lds_off32(&Qs[0][0]);
    const unsigned kbase = lds_off32(&Ks[0][0]);
    const unsigned vbase = lds_off32(&Vs[0][0]);

    // ---- async-load Q tile, build A fragments for all 4 d-chunks ----
    async_load_tile(Qg + headoff + (size_t)qt * 64 * HD, qbase, tid);
    wait_async_all();
    __syncthreads();

    v16bf qf[4];
    #pragma unroll
    for (int dc = 0; dc < 4; ++dc)
        #pragma unroll
        for (int hh = 0; hh < 16; ++hh) {
            int v  = hh >> 1;
            int kk = ((v < 4) ? 0 : 16) + (hi ? 8 : 0) + ((v & 3) << 1) + (hh & 1);
            qf[dc][hh] = Qs[wave * 16 + ln][dc * 32 + kk];
        }

    float m_i[8], l_i[8];
    v8f of[8];
    #pragma unroll
    for (int r = 0; r < 8; ++r) { m_i[r] = -1e30f; l_i[r] = 0.f; }
    #pragma unroll
    for (int j = 0; j < 8; ++j)
        of[j] = (v8f){0.f,0.f,0.f,0.f,0.f,0.f,0.f,0.f};

    const float scale = 0.08838834764831845f;   // 1/sqrt(128)

    for (int t = 0; t <= qt; ++t) {
        __syncthreads();
        async_load_tile(Kg + headoff + (size_t)t * 64 * HD, kbase, tid);
        async_load_tile(Vg + headoff + (size_t)t * 64 * HD, vbase, tid);
        wait_async_all();
        __syncthreads();
        if (t < qt)   // hint next K tile into cache (global_prefetch_b8)
            __builtin_prefetch(Kg + headoff + (size_t)(t + 1) * 64 * HD + tid * 64, 0, 1);

        // ---- S = Q K^T  (4 kv-subtiles x 4 d-chunks) ----
        v8f sf[4];
        #pragma unroll
        for (int sub = 0; sub < 4; ++sub)
            sf[sub] = (v8f){0.f,0.f,0.f,0.f,0.f,0.f,0.f,0.f};
        #pragma unroll
        for (int dc = 0; dc < 4; ++dc)
            #pragma unroll
            for (int sub = 0; sub < 4; ++sub) {
                v16bf bm;   // B[d][k] = K[kpos][d]
                #pragma unroll
                for (int hh = 0; hh < 16; ++hh)
                    bm[hh] = Ks[sub * 16 + ln][dc * 32 + ((hh & 8) << 1) + (hi ? 8 : 0) + (hh & 7)];
                sf[sub] = __builtin_amdgcn_wmma_f32_16x16x32_bf16(
                    false, qf[dc], false, bm, (short)0, sf[sub], false, false);
            }

        // ---- scale + causal mask on diagonal tile ----
        #pragma unroll
        for (int sub = 0; sub < 4; ++sub)
            #pragma unroll
            for (int r = 0; r < 8; ++r) {
                float v = sf[sub][r] * scale;
                if (t == qt) {
                    int kgl = t * 64 + sub * 16 + ln;
                    int qgl = qt * 64 + wave * 16 + r + hi * 8;
                    if (kgl > qgl) v = -1e9f;
                }
                sf[sub][r] = v;
            }

        // ---- online softmax: row reductions over 16 lanes of same half ----
        float mnew[8], alpha[8];
        #pragma unroll
        for (int r = 0; r < 8; ++r) {
            float mx = fmaxf(fmaxf(sf[0][r], sf[1][r]), fmaxf(sf[2][r], sf[3][r]));
            #pragma unroll
            for (int m = 1; m <= 8; m <<= 1) mx = fmaxf(mx, __shfl_xor(mx, m, 32));
            mnew[r]  = fmaxf(m_i[r], mx);
            alpha[r] = __expf(m_i[r] - mnew[r]);
        }
        #pragma unroll
        for (int r = 0; r < 8; ++r) {
            float rs = 0.f;
            #pragma unroll
            for (int sub = 0; sub < 4; ++sub) {
                float p = __expf(sf[sub][r] - mnew[r]);
                sf[sub][r] = p;
                rs += p;
            }
            #pragma unroll
            for (int m = 1; m <= 8; m <<= 1) rs += __shfl_xor(rs, m, 32);
            l_i[r] = l_i[r] * alpha[r] + rs;
            m_i[r] = mnew[r];
        }
        #pragma unroll
        for (int j = 0; j < 8; ++j)
            #pragma unroll
            for (int r = 0; r < 8; ++r) of[j][r] *= alpha[r];

        // ---- P: C-layout -> A-layout via per-wave LDS staging ----
        #pragma unroll
        for (int sub = 0; sub < 4; ++sub)
            #pragma unroll
            for (int r = 0; r < 8; ++r)
                Ps[wave][r + hi * 8][sub * 16 + ln] = f2bf(sf[sub][r]);

        // ---- O += P V  (2 kv-chunks of 32 x 8 d-subtiles) ----
        #pragma unroll
        for (int kc2 = 0; kc2 < 2; ++kc2) {
            v16bf pf;
            #pragma unroll
            for (int hh = 0; hh < 16; ++hh) {
                int v  = hh >> 1;
                int kk = ((v < 4) ? 0 : 16) + (hi ? 8 : 0) + ((v & 3) << 1) + (hh & 1);
                pf[hh] = Ps[wave][ln][kc2 * 32 + kk];
            }
            #pragma unroll
            for (int j = 0; j < 8; ++j) {
                v16bf bm;   // B[kv][d] = V[kvpos][d]
                #pragma unroll
                for (int hh = 0; hh < 16; ++hh)
                    bm[hh] = Vs[kc2 * 32 + ((hh & 8) << 1) + (hi ? 8 : 0) + (hh & 7)][j * 16 + ln];
                of[j] = __builtin_amdgcn_wmma_f32_16x16x32_bf16(
                    false, pf, false, bm, (short)0, of[j], false, false);
            }
        }
    }

    // ---- normalize + store f32 out[b][s][h*128 + d] ----
    #pragma unroll
    for (int j = 0; j < 8; ++j)
        #pragma unroll
        for (int r = 0; r < 8; ++r) {
            int sgl = qt * 64 + wave * 16 + r + hi * 8;
            float val = of[j][r] / l_i[r];
            out[((size_t)(b * Sq + sgl)) * DIMe + h * HD + j * 16 + ln] = val;
        }
}

// ---------------------------------------------------------------------------
// Host launcher. Inputs: x, mask(unused - causal applied analytically),
// freqs, Wq, Wk, Wv. Workspace: Q,K,V bf16 [b][h][s][d] = 3 * 16 MB.
// ---------------------------------------------------------------------------
extern "C" void kernel_launch(void* const* d_in, const int* in_sizes, int n_in,
                              void* d_out, int out_size, void* d_ws, size_t ws_size,
                              hipStream_t stream) {
    const float* x     = (const float*)d_in[0];
    const float* freqs = (const float*)d_in[2];
    const float* Wq    = (const float*)d_in[3];
    const float* Wk    = (const float*)d_in[4];
    const float* Wv    = (const float*)d_in[5];
    (void)in_sizes; (void)n_in; (void)out_size; (void)ws_size;

    const size_t QKV_ELEMS = (size_t)Bz * NH * Sq * HD;   // 8,388,608
    __bf16* q = (__bf16*)d_ws;
    __bf16* k = q + QKV_ELEMS;
    __bf16* v = k + QKV_ELEMS;

    dim3 g1(DIMe / 64, (Bz * Sq) / 64, 3);
    qkv_rope_kernel<<<g1, 128, 0, stream>>>(x, Wq, Wk, Wv, freqs, q, k, v);

    dim3 g2(Sq / 64, NH, Bz);
    flash_attn_kernel<<<g2, 128, 0, stream>>>(q, k, v, (float*)d_out);
}
